// TransformerDecoderLayer_26826365731387
// MI455X (gfx1250) — matrix operand order repaired and checked by reading, not compiled, not run on hardware
//
#include <hip/hip_runtime.h>

typedef __attribute__((ext_vector_type(16))) __bf16 bf16x16;
typedef __attribute__((ext_vector_type(8)))  float  f32x8;

union FragAB {
  bf16x16 v;
  uint4 q[2];
  __bf16 h[16];
};

__device__ __forceinline__ f32x8 wmma_bf16(bf16x16 a, bf16x16 b, f32x8 c) {
  // D = A(16x32 bf16) * B(32x16 bf16) + C(16x16 f32)
  return __builtin_amdgcn_wmma_f32_16x16x32_bf16(false, a, false, b, (short)0, c,
                                                 false, false);
}

__device__ __forceinline__ void cvt8(const float* p, __bf16* d) {
  float4 f0 = *(const float4*)p;
  float4 f1 = *(const float4*)(p + 4);
  d[0] = (__bf16)f0.x; d[1] = (__bf16)f0.y; d[2] = (__bf16)f0.z; d[3] = (__bf16)f0.w;
  d[4] = (__bf16)f1.x; d[5] = (__bf16)f1.y; d[6] = (__bf16)f1.z; d[7] = (__bf16)f1.w;
}
__device__ __forceinline__ void cvt8(const __bf16* p, __bf16* d) {
  *(uint4*)d = *(const uint4*)p;
}

// Async global->LDS copy of 16 bytes per lane (CDNA5 ASYNCcnt path).
__device__ __forceinline__ void async_copy_b128(const __bf16* gptr, __bf16* lptr) {
  unsigned lo = (unsigned)(uintptr_t)lptr;                 // LDS offset (low 32b)
  unsigned long long ga = (unsigned long long)(uintptr_t)gptr;
  asm volatile("global_load_async_to_lds_b128 %0, %1, off" ::"v"(lo), "v"(ga)
               : "memory");
}
__device__ __forceinline__ void wait_async() {
  asm volatile("s_wait_asynccnt 0" ::: "memory");
}

// ---------------------------------------------------------------------------
// WMMA GEMM: C[M,N] = A[M,K] * Wt[N,K]^T + bias. Block tile 128x128, BK=32,
// double-buffered LDS. 8 waves; each wave computes 32x64 via 2x4 WMMA tiles.
// A (f32 or bf16) staged via VGPRs with convert; W tile streamed with
// global_load_async_to_lds_b128 (ASYNCcnt) straight into LDS.
// ---------------------------------------------------------------------------
template <typename AT, bool RELU, bool OUT_BF16>
__global__ __launch_bounds__(256) void gemm_wmma(
    const AT* __restrict__ A, const __bf16* __restrict__ Wt,
    const float* __restrict__ bias, void* __restrict__ Cout,
    int M, int N, int K) {
  __shared__ alignas(16) __bf16 As[2][128 * 32];
  __shared__ alignas(16) __bf16 Bs[2][128 * 32];

  const int tid  = threadIdx.x;
  const int lane = tid & 31;
  const int wave = tid >> 5;
  const int wm = wave >> 1;        // 0..3 : 32-row stripe
  const int wn = wave & 1;         // 0..1 : 64-col stripe
  const int l16  = lane & 15;
  const int half = lane >> 4;
  const int bm = blockIdx.y * 128;
  const int bn = blockIdx.x * 128;

  f32x8 acc[2][4];
#pragma unroll
  for (int i = 0; i < 2; ++i)
#pragma unroll
    for (int j = 0; j < 4; ++j)
#pragma unroll
      for (int e = 0; e < 8; ++e) acc[i][j][e] = 0.f;

  // Stage first K-tile into buffer 0.
#pragma unroll
  for (int it = 0; it < 2; ++it) {
    int seg = tid + it * 256;          // 0..511
    int row = seg >> 2;                // 0..127
    int kc  = (seg & 3) << 3;          // 0,8,16,24
    __bf16 t8[8];
    cvt8(A + (size_t)(bm + row) * K + kc, t8);
    *(uint4*)&As[0][row * 32 + kc] = *(uint4*)t8;
  }
#pragma unroll
  for (int it = 0; it < 2; ++it) {
    int seg = tid + it * 256;
    int row = seg >> 2;
    int kc  = (seg & 3) << 3;
    async_copy_b128(Wt + (size_t)(bn + row) * K + kc, &Bs[0][row * 32 + kc]);
  }
  wait_async();
  __syncthreads();

  for (int k0 = 0; k0 < K; k0 += 32) {
    const int cur = (k0 >> 5) & 1;
    const int nxt = cur ^ 1;
    const bool more = (k0 + 32) < K;

    // Prefetch next tile: A into registers, W async into next LDS buffer.
    __bf16 t8a[2][8];
    if (more) {
#pragma unroll
      for (int it = 0; it < 2; ++it) {
        int seg = tid + it * 256;
        int row = seg >> 2;
        int kc  = (seg & 3) << 3;
        cvt8(A + (size_t)(bm + row) * K + (k0 + 32) + kc, t8a[it]);
      }
#pragma unroll
      for (int it = 0; it < 2; ++it) {
        int seg = tid + it * 256;
        int row = seg >> 2;
        int kc  = (seg & 3) << 3;
        async_copy_b128(Wt + (size_t)(bn + row) * K + (k0 + 32) + kc,
                        &Bs[nxt][row * 32 + kc]);
      }
    }

    // Fragments from current buffer.
    // A frag: lane=row, elems 0..7 -> K=half*8+0..7, 8..15 -> K=16+half*8+0..7.
    FragAB af[2], bfr[4];
#pragma unroll
    for (int i = 0; i < 2; ++i) {
      const __bf16* p = &As[cur][(wm * 32 + i * 16 + l16) * 32];
      af[i].q[0] = *(const uint4*)(p + half * 8);
      af[i].q[1] = *(const uint4*)(p + 16 + half * 8);
    }
    // B frag: lane=col, elems 0..7 -> K=half*8+0..7, 8..15 -> K=16+half*8+0..7.
#pragma unroll
    for (int j = 0; j < 4; ++j) {
      const __bf16* p = &Bs[cur][(wn * 64 + j * 16 + l16) * 32];
      bfr[j].q[0] = *(const uint4*)(p + half * 8);
      bfr[j].q[1] = *(const uint4*)(p + 16 + half * 8);
    }
#pragma unroll
    for (int i = 0; i < 2; ++i)
#pragma unroll
      for (int j = 0; j < 4; ++j)
        acc[i][j] = wmma_bf16(af[i].v, bfr[j].v, acc[i][j]);

    if (more) {
#pragma unroll
      for (int it = 0; it < 2; ++it) {
        int seg = tid + it * 256;
        int row = seg >> 2;
        int kc  = (seg & 3) << 3;
        *(uint4*)&As[nxt][row * 32 + kc] = *(uint4*)t8a[it];
      }
      wait_async();
    }
    __syncthreads();
  }

  // D layout: VGPR r -> row M = r + 8*half; col N = l16.
#pragma unroll
  for (int i = 0; i < 2; ++i) {
    int rowb = bm + wm * 32 + i * 16 + half * 8;
#pragma unroll
    for (int j = 0; j < 4; ++j) {
      int col = bn + wn * 64 + j * 16 + l16;
      float bv = bias[col];
#pragma unroll
      for (int r = 0; r < 8; ++r) {
        float vv = acc[i][j][r] + bv;
        if (RELU) vv = fmaxf(vv, 0.f);
        size_t o = (size_t)(rowb + r) * N + col;
        if (OUT_BF16) ((__bf16*)Cout)[o] = (__bf16)vv;
        else          ((float*)Cout)[o]  = vv;
      }
    }
  }
}

// ---------------------------------------------------------------------------
// Flash attention: one wave per (batch, head, 16-row query tile).
// Q,K bf16 [token, D]; V pre-transposed bf16 [b, h, dh, t]; O f32 [token, D].
// Online softmax done lane-locally after an LDS transpose of the score tile
// (no ds_bpermute chains); P fragment built directly in registers.
// ---------------------------------------------------------------------------
__global__ __launch_bounds__(128) void flash_attn(
    const __bf16* __restrict__ Q, const __bf16* __restrict__ Kb,
    const __bf16* __restrict__ Vt, float* __restrict__ O, int Tq, int Tk) {
  constexpr int SROW = 36;  // padded f32 row stride (words) to spread banks
  __shared__ alignas(16) float sbuf[4][16 * SROW];
  __shared__ alignas(16) float rbuf[4][16];

  const int lane = threadIdx.x & 31;
  const int w    = threadIdx.x >> 5;
  const int l16  = lane & 15;
  const int half = lane >> 4;
  const int D = 1024, H = 16;

  int idx    = blockIdx.x * 4 + w;
  int qtiles = Tq >> 4;
  int qt = idx % qtiles;
  int h  = (idx / qtiles) % H;
  int b  = idx / (qtiles * H);

  // Q fragments (A layout), dh=64 -> two K-chunks of 32.
  FragAB qf[2];
  {
    const __bf16* qp = Q + (size_t)(b * Tq + qt * 16 + l16) * D + h * 64;
#pragma unroll
    for (int c = 0; c < 2; ++c) {
      qf[c].q[0] = *(const uint4*)(qp + c * 32 + half * 8);
      qf[c].q[1] = *(const uint4*)(qp + c * 32 + 16 + half * 8);
    }
  }

  float mrow = -1e30f, lrow = 0.f;  // state for row l16 (both halves redundant)
  f32x8 acc[4];
#pragma unroll
  for (int n = 0; n < 4; ++n)
#pragma unroll
    for (int e = 0; e < 8; ++e) acc[n][e] = 0.f;

  for (int kt = 0; kt < Tk; kt += 32) {
    // --- load all K fragments, then the score WMMAs ---
    FragAB kf[2][2];
#pragma unroll
    for (int g = 0; g < 2; ++g) {
      const __bf16* kp = Kb + (size_t)(b * Tk + kt + g * 16 + l16) * D + h * 64;
#pragma unroll
      for (int c = 0; c < 2; ++c) {
        kf[g][c].q[0] = *(const uint4*)(kp + c * 32 + half * 8);
        kf[g][c].q[1] = *(const uint4*)(kp + c * 32 + 16 + half * 8);
      }
    }
    f32x8 s[2];
#pragma unroll
    for (int g = 0; g < 2; ++g) {
#pragma unroll
      for (int e = 0; e < 8; ++e) s[g][e] = 0.f;
#pragma unroll
      for (int c = 0; c < 2; ++c) s[g] = wmma_bf16(qf[c].v, kf[g][c].v, s[g]);
    }

    // V fragments early (independent of softmax; hides global latency).
    FragAB vf[4];
#pragma unroll
    for (int n = 0; n < 4; ++n) {
      const __bf16* vp =
          Vt + (size_t)((b * H + h) * 64 + n * 16 + l16) * Tk + kt;
      vf[n].q[0] = *(const uint4*)(vp + half * 8);
      vf[n].q[1] = *(const uint4*)(vp + 16 + half * 8);
    }

    // --- transpose scores via LDS: D layout -> each lane owns row l16 ---
    __syncthreads();
#pragma unroll
    for (int r = 0; r < 8; ++r) {
      int row = r + half * 8;
      sbuf[w][row * SROW + l16]      = s[0][r] * 0.125f;  // 1/sqrt(64)
      sbuf[w][row * SROW + 16 + l16] = s[1][r] * 0.125f;
    }
    __syncthreads();

    // --- lane-local online softmax over the full 32-wide row ---
    float e[32];
    {
      const float* sr = &sbuf[w][l16 * SROW];
      float4 rv[8];
#pragma unroll
      for (int i = 0; i < 8; ++i) rv[i] = *(const float4*)(sr + i * 4);
      float mx = -1e30f;
#pragma unroll
      for (int i = 0; i < 8; ++i)
        mx = fmaxf(mx, fmaxf(fmaxf(rv[i].x, rv[i].y), fmaxf(rv[i].z, rv[i].w)));
      float mn  = fmaxf(mrow, mx);
      float cor = __expf(mrow - mn);
      float rs  = 0.f;
#pragma unroll
      for (int i = 0; i < 8; ++i) {
        e[i * 4 + 0] = __expf(rv[i].x - mn);
        e[i * 4 + 1] = __expf(rv[i].y - mn);
        e[i * 4 + 2] = __expf(rv[i].z - mn);
        e[i * 4 + 3] = __expf(rv[i].w - mn);
        rs += e[i * 4 + 0] + e[i * 4 + 1] + e[i * 4 + 2] + e[i * 4 + 3];
      }
      lrow = lrow * cor + rs;
      mrow = mn;
      if (half == 0) rbuf[w][l16] = cor;  // broadcast per-row correction
    }
    __syncthreads();

    // Rescale accumulators: rows r + 8*half.
    {
      float4 c0 = *(const float4*)&rbuf[w][half * 8];
      float4 c1 = *(const float4*)&rbuf[w][half * 8 + 4];
      float corr[8] = {c0.x, c0.y, c0.z, c0.w, c1.x, c1.y, c1.z, c1.w};
#pragma unroll
      for (int n = 0; n < 4; ++n)
#pragma unroll
        for (int r = 0; r < 8; ++r) acc[n][r] *= corr[r];
    }

    // Build P fragment in registers (A layout: elem i -> K = 16*blk + half*8+i).
    FragAB pf;
#pragma unroll
    for (int i = 0; i < 8; ++i) {
      pf.h[i]     = (__bf16)(half ? e[8 + i]  : e[i]);
      pf.h[8 + i] = (__bf16)(half ? e[24 + i] : e[16 + i]);
    }

    // O += P(16x32) * V(32x64).
#pragma unroll
    for (int n = 0; n < 4; ++n) acc[n] = wmma_bf16(pf.v, vf[n].v, acc[n]);
  }

  // Broadcast 1/l per row, then final scaled store.
  __syncthreads();
  if (half == 0) rbuf[w][l16] = 1.f / lrow;
  __syncthreads();
  float4 l0 = *(const float4*)&rbuf[w][half * 8];
  float4 l1 = *(const float4*)&rbuf[w][half * 8 + 4];
  float linv[8] = {l0.x, l0.y, l0.z, l0.w, l1.x, l1.y, l1.z, l1.w};
#pragma unroll
  for (int n = 0; n < 4; ++n) {
    int col = h * 64 + n * 16 + l16;
#pragma unroll
    for (int r = 0; r < 8; ++r) {
      int row = qt * 16 + r + half * 8;
      O[(size_t)(b * Tq + row) * D + col] = acc[n][r] * linv[r];
    }
  }
}

// ---------------------------------------------------------------------------
// out = LayerNorm(X + Y) * g + be ; one 256-thread block per 1024-wide row.
// ---------------------------------------------------------------------------
__global__ __launch_bounds__(256) void add_ln(
    const float* __restrict__ X, const float* __restrict__ Y,
    const float* __restrict__ g, const float* __restrict__ be,
    float* __restrict__ out) {
  __shared__ float red[2][8];
  const int row = blockIdx.x;
  const int tid = threadIdx.x;
  const int D = 1024;
  const float* xr = X + (size_t)row * D + tid * 4;
  const float* yr = Y + (size_t)row * D + tid * 4;
  float4 a = *(const float4*)xr;
  float4 b = *(const float4*)yr;
  float v[4] = {a.x + b.x, a.y + b.y, a.z + b.z, a.w + b.w};
  float s = 0.f, s2 = 0.f;
#pragma unroll
  for (int i = 0; i < 4; ++i) { s += v[i]; s2 += v[i] * v[i]; }
#pragma unroll
  for (int o = 1; o < 32; o <<= 1) {
    s  += __shfl_xor(s, o, 32);
    s2 += __shfl_xor(s2, o, 32);
  }
  int wv = tid >> 5, ln = tid & 31;
  if (ln == 0) { red[0][wv] = s; red[1][wv] = s2; }
  __syncthreads();
  if (wv == 0) {
    float t  = (ln < 8) ? red[0][ln] : 0.f;
    float t2 = (ln < 8) ? red[1][ln] : 0.f;
#pragma unroll
    for (int o = 1; o < 8; o <<= 1) {
      t  += __shfl_xor(t, o, 32);
      t2 += __shfl_xor(t2, o, 32);
    }
    if (ln == 0) { red[0][0] = t; red[1][0] = t2; }
  }
  __syncthreads();
  float mu  = red[0][0] * (1.f / 1024.f);
  float var = red[1][0] * (1.f / 1024.f) - mu * mu;
  float inv = rsqrtf(var + 1e-5f);
  float* op = out + (size_t)row * D + tid * 4;
#pragma unroll
  for (int i = 0; i < 4; ++i)
    op[i] = (v[i] - mu) * inv * g[tid * 4 + i] + be[tid * 4 + i];
}

// Transpose-convert weight f32 [K,N] -> bf16 [N,K].
__global__ void convert_wt(const float* __restrict__ w, __bf16* __restrict__ wt,
                           int K, int N) {
  size_t idx = (size_t)blockIdx.x * 256 + threadIdx.x;
  if (idx >= (size_t)K * N) return;
  size_t n = idx / K, k = idx % K;
  wt[idx] = (__bf16)w[k * N + n];
}

// Transpose V bf16 [b*T + t, 1024] -> [(b*16+h)*64+dh, T]
__global__ void transpose_v(const __bf16* __restrict__ v,
                            __bf16* __restrict__ vt, int T) {
  size_t idx = (size_t)blockIdx.x * 256 + threadIdx.x;  // total 8*16*64*T
  int t = (int)(idx % T);
  size_t rest = idx / T;  // (b*16+h)*64 + dh
  int dh = (int)(rest % 64);
  int bh = (int)(rest / 64);
  int h = bh % 16, b = bh / 16;
  vt[idx] = v[((size_t)b * T + t) * 1024 + h * 64 + dh];
}

extern "C" void kernel_launch(void* const* d_in, const int* in_sizes, int n_in,
                              void* d_out, int out_size, void* d_ws,
                              size_t ws_size, hipStream_t stream) {
  (void)in_sizes; (void)n_in; (void)out_size; (void)ws_size;
  const float* x     = (const float*)d_in[0];
  const float* enc   = (const float*)d_in[1];
  const float* sa_wq = (const float*)d_in[2];  const float* sa_bq = (const float*)d_in[3];
  const float* sa_wk = (const float*)d_in[4];  const float* sa_bk = (const float*)d_in[5];
  const float* sa_wv = (const float*)d_in[6];  const float* sa_bv = (const float*)d_in[7];
  const float* sa_wo = (const float*)d_in[8];  const float* sa_bo = (const float*)d_in[9];
  const float* ca_wq = (const float*)d_in[10]; const float* ca_bq = (const float*)d_in[11];
  const float* ca_wk = (const float*)d_in[12]; const float* ca_bk = (const float*)d_in[13];
  const float* ca_wv = (const float*)d_in[14]; const float* ca_bv = (const float*)d_in[15];
  const float* ca_wo = (const float*)d_in[16]; const float* ca_bo = (const float*)d_in[17];
  const float* w1    = (const float*)d_in[18]; const float* b1    = (const float*)d_in[19];
  const float* w2    = (const float*)d_in[20]; const float* b2    = (const float*)d_in[21];
  const float* g1    = (const float*)d_in[22]; const float* be1   = (const float*)d_in[23];
  const float* g2    = (const float*)d_in[24]; const float* be2   = (const float*)d_in[25];
  const float* g3    = (const float*)d_in[26]; const float* be3   = (const float*)d_in[27];

  const int D = 1024, F = 4096, T = 1024;
  const int M = 8 * T;  // 8192 tokens

  char* ws = (char*)d_ws;
  size_t off = 0;
  auto take = [&](size_t bytes) -> void* {
    void* p = ws + off;
    off += (bytes + 255) & ~(size_t)255;
    return p;
  };

  __bf16* wt_saq = (__bf16*)take((size_t)D * D * 2);
  __bf16* wt_sak = (__bf16*)take((size_t)D * D * 2);
  __bf16* wt_sav = (__bf16*)take((size_t)D * D * 2);
  __bf16* wt_sao = (__bf16*)take((size_t)D * D * 2);
  __bf16* wt_caq = (__bf16*)take((size_t)D * D * 2);
  __bf16* wt_cak = (__bf16*)take((size_t)D * D * 2);
  __bf16* wt_cav = (__bf16*)take((size_t)D * D * 2);
  __bf16* wt_cao = (__bf16*)take((size_t)D * D * 2);
  __bf16* w1t    = (__bf16*)take((size_t)D * F * 2);  // [F, D]
  __bf16* w2t    = (__bf16*)take((size_t)F * D * 2);  // [D, F]
  __bf16* qb = (__bf16*)take((size_t)M * D * 2);
  __bf16* kb = (__bf16*)take((size_t)M * D * 2);
  __bf16* vb = (__bf16*)take((size_t)M * D * 2);
  __bf16* vt = (__bf16*)take((size_t)M * D * 2);
  __bf16* ff1 = qb;  // reuse retired q/k/v/vt region (64 MB) for FF activation
  float* attn = (float*)take((size_t)M * D * 4);
  float* tmp  = (float*)take((size_t)M * D * 4);
  float* h1   = (float*)take((size_t)M * D * 4);
  float* h2   = (float*)take((size_t)M * D * 4);

  const int wgrid_DD = (D * D + 255) / 256;
  const int wgrid_DF = (D * F + 255) / 256;
  convert_wt<<<wgrid_DD, 256, 0, stream>>>(sa_wq, wt_saq, D, D);
  convert_wt<<<wgrid_DD, 256, 0, stream>>>(sa_wk, wt_sak, D, D);
  convert_wt<<<wgrid_DD, 256, 0, stream>>>(sa_wv, wt_sav, D, D);
  convert_wt<<<wgrid_DD, 256, 0, stream>>>(sa_wo, wt_sao, D, D);
  convert_wt<<<wgrid_DD, 256, 0, stream>>>(ca_wq, wt_caq, D, D);
  convert_wt<<<wgrid_DD, 256, 0, stream>>>(ca_wk, wt_cak, D, D);
  convert_wt<<<wgrid_DD, 256, 0, stream>>>(ca_wv, wt_cav, D, D);
  convert_wt<<<wgrid_DD, 256, 0, stream>>>(ca_wo, wt_cao, D, D);
  convert_wt<<<wgrid_DF, 256, 0, stream>>>(w1, w1t, D, F);
  convert_wt<<<wgrid_DF, 256, 0, stream>>>(w2, w2t, F, D);

  dim3 gN1024(D / 128, M / 128);  // 8 x 64
  dim3 gN4096(F / 128, M / 128);  // 32 x 64
  const int vtgrid = (M * D + 255) / 256;
  const int fgrid  = 8 * 16 * (T / 16) / 4;  // 2048 blocks, 4 waves each

  // --- self attention ---
  gemm_wmma<float, false, true><<<gN1024, 256, 0, stream>>>(x, wt_saq, sa_bq, qb, M, D, D);
  gemm_wmma<float, false, true><<<gN1024, 256, 0, stream>>>(x, wt_sak, sa_bk, kb, M, D, D);
  gemm_wmma<float, false, true><<<gN1024, 256, 0, stream>>>(x, wt_sav, sa_bv, vb, M, D, D);
  transpose_v<<<vtgrid, 256, 0, stream>>>(vb, vt, T);
  flash_attn<<<fgrid, 128, 0, stream>>>(qb, kb, vt, attn, T, T);
  gemm_wmma<float, false, false><<<gN1024, 256, 0, stream>>>(attn, wt_sao, sa_bo, tmp, M, D, D);
  add_ln<<<M, 256, 0, stream>>>(x, tmp, g1, be1, h1);

  // --- cross attention ---
  gemm_wmma<float, false, true><<<gN1024, 256, 0, stream>>>(h1, wt_caq, ca_bq, qb, M, D, D);
  gemm_wmma<float, false, true><<<gN1024, 256, 0, stream>>>(enc, wt_cak, ca_bk, kb, M, D, D);
  gemm_wmma<float, false, true><<<gN1024, 256, 0, stream>>>(enc, wt_cav, ca_bv, vb, M, D, D);
  transpose_v<<<vtgrid, 256, 0, stream>>>(vb, vt, T);
  flash_attn<<<fgrid, 128, 0, stream>>>(qb, kb, vt, attn, T, T);
  gemm_wmma<float, false, false><<<gN1024, 256, 0, stream>>>(attn, wt_cao, ca_bo, tmp, M, D, D);
  add_ln<<<M, 256, 0, stream>>>(h1, tmp, g2, be2, h2);

  // --- feed forward ---
  gemm_wmma<float, true, true><<<gN4096, 256, 0, stream>>>(h2, w1t, b1, ff1, M, F, D);
  gemm_wmma<__bf16, false, false><<<gN1024, 256, 0, stream>>>(ff1, w2t, b2, tmp, M, D, F);
  add_ln<<<M, 256, 0, stream>>>(h2, tmp, g3, be3, (float*)d_out);
}
